// NoisyActLin_60009283059958
// MI455X (gfx1250) — matrix operand order repaired and checked by reading, not compiled, not run on hardware
//
#include <hip/hip_runtime.h>

// ---------------------------------------------------------------------------
// NoisyActLin for MI455X (gfx1250, wave32, WMMA + Tensor Data Mover).
//  M = B*S = 8192, N = DOUT = 8192, K = DIN = 2048
//  Stage 1: fake-quant activations -> f16   (d_ws, 32 MiB)
//  Stage 2: fake-quant weights     -> f16   (d_ws + 32 MiB)
//  Stage 3: f16 WMMA GEMM, TDM double-buffered LDS staging, f32 accum + bias
// Requires ws_size >= 64 MiB.
// ---------------------------------------------------------------------------

typedef __attribute__((ext_vector_type(8)))  _Float16     v8h;
typedef __attribute__((ext_vector_type(16))) _Float16     v16h;
typedef __attribute__((ext_vector_type(8)))  float        v8f;
typedef __attribute__((ext_vector_type(4)))  unsigned int v4u;
typedef __attribute__((ext_vector_type(8)))  int          v8i;
typedef __attribute__((ext_vector_type(4)))  int          v4i;

#define DIN   2048
#define DOUT  8192
#define MTOT  8192          // B * S = 4 * 2048
#define BM    128
#define BN    128
#define BK    32
#define LDSS  40            // 32 data halves + 8 pad halves -> 80B row stride

union Frag16 { v16h v; v8h h[2]; };

// ---------------- Stage 1: activation fake-quant (per-tensor) ----------------
__global__ __launch_bounds__(256)
void quant_act(const float* __restrict__ x, _Float16* __restrict__ xq,
               const float* __restrict__ log_act_s, const float* __restrict__ log_act_q,
               const float* __restrict__ act_b, int n) {
  const float s    = exp2f(log_act_s[0]);
  const float q    = exp2f(log_act_q[0]);
  const float zp   = rintf(act_b[0] / s * 2.0f) * 0.5f * s;  // guard-bit rounded zero point
  const float lo   = zp;
  const float hi   = zp + q - s;
  const float invs = 1.0f / s;                               // s is a power of two: exact

  const int stride = gridDim.x * blockDim.x;
  for (int i = blockIdx.x * blockDim.x + threadIdx.x; i * 8 < n; i += stride) {
    const float4* p = (const float4*)(x + (size_t)i * 8);
    float4 v0 = p[0];
    float4 v1 = p[1];
    float vv[8] = {v0.x, v0.y, v0.z, v0.w, v1.x, v1.y, v1.z, v1.w};
    v8h o;
#pragma unroll
    for (int j = 0; j < 8; ++j) {
      float c  = fminf(fmaxf(vv[j], lo), hi);
      float qv = rintf((c - zp) * invs);       // STE round == plain RNE round in fwd
      o[j] = (_Float16)(qv * s + zp);
    }
    *(v8h*)(xq + (size_t)i * 8) = o;
  }
}

// ------------- Stage 2: weight fake-quant (per-out-channel min/max) ----------
__global__ __launch_bounds__(256)
void quant_wgt(const float* __restrict__ w, _Float16* __restrict__ wq,
               const float* __restrict__ log_wght_s) {
  __shared__ float smin[256];
  __shared__ float smax[256];
  const int o = blockIdx.x;          // output channel (row of W), 8192 blocks
  const int t = threadIdx.x;         // 256 threads * 8 elements = DIN
  const float* wr = w + (size_t)o * DIN;

  float4 a  = ((const float4*)wr)[t * 2 + 0];
  float4 b4 = ((const float4*)wr)[t * 2 + 1];
  float v[8] = {a.x, a.y, a.z, a.w, b4.x, b4.y, b4.z, b4.w};

  float mn = v[0], mx = v[0];
#pragma unroll
  for (int j = 1; j < 8; ++j) { mn = fminf(mn, v[j]); mx = fmaxf(mx, v[j]); }
  smin[t] = mn; smax[t] = mx;
  __syncthreads();
  for (int off = 128; off > 0; off >>= 1) {
    if (t < off) {
      smin[t] = fminf(smin[t], smin[t + off]);
      smax[t] = fmaxf(smax[t], smax[t + off]);
    }
    __syncthreads();
  }

  const float ws    = exp2f(log_wght_s[o]);
  const float qwmin = rintf(smin[0] / ws * 2.0f) * 0.5f * ws;
  const float qwmax = rintf(smax[0] / ws * 2.0f) * 0.5f * ws;
  const float invs  = 1.0f / ws;

  v8h o8;
#pragma unroll
  for (int j = 0; j < 8; ++j) {
    float c  = fminf(fmaxf(v[j], qwmin), qwmax);
    float qv = rintf((c - qwmin) * invs);
    o8[j] = (_Float16)(qv * ws + qwmin);
  }
  *(v8h*)(wq + (size_t)o * DIN + t * 8) = o8;
}

// ---------------- TDM: DMA one 128x32 f16 tile (row-major) into LDS ----------
// LDS layout matches compute: row stride 80B = 16 dwords data + 4 dwords pad.
__device__ __forceinline__
void tdm_load_tile(const _Float16* gtile, unsigned lds_byte, int rows_total) {
  unsigned long long ga = (unsigned long long)(size_t)gtile;
  v4u g0;
  g0[0] = 1u;                                   // count=1, user descriptor
  g0[1] = lds_byte;                             // D#.lds_addr
  g0[2] = (unsigned)(ga & 0xffffffffu);         // global_addr[31:0]
  g0[3] = (unsigned)((ga >> 32) & 0x01ffffffu)  // global_addr[56:32]
        | (2u << 30);                           // type = 2 ("image")
  v8i g1;
  g1[0] = (1 << 16)                             // data_size = 2 bytes
        | (1 << 20)                             // pad_enable
        | (3 << 22)                             // pad_interval: 16 dwords (64B)
        | (3 << 25);                            // pad_amount:   4 dwords (16B)
  g1[1] = (int)(((unsigned)DIN & 0xffffu) << 16);            // tensor_dim0[15:0]
  g1[2] = (int)(((unsigned)DIN >> 16) |
                (((unsigned)rows_total & 0xffffu) << 16));   // dim0 hi | dim1 lo
  g1[3] = (int)(((unsigned)rows_total >> 16) |
                ((unsigned)BK << 16));                       // dim1 hi | tile_dim0=32
  g1[4] = BM;                                   // tile_dim1 = 128, tile_dim2 = 0
  g1[5] = DIN;                                  // tensor_dim0_stride[31:0] (elements)
  g1[6] = 0;                                    // stride hi | dim1_stride lo (unused, 2D)
  g1[7] = 0;
  v4i z4 = {0, 0, 0, 0};                        // groups 2/3 unused (2D tile)
  v8i z8 = {0, 0, 0, 0, 0, 0, 0, 0};            // trailing group (unused)
  __builtin_amdgcn_tensor_load_to_lds(g0, g1, z4, z4, z8, 0);
}

// ------------------- Stage 3: f16 WMMA GEMM  y = Xq @ Wq^T + b ---------------
// Block tile 128x128, BK=32, TDM double-buffered LDS. 8 waves as 2(M) x 4(N);
// wave tile 64x32 -> 4x2 accumulators, 8 x v_wmma_f32_16x16x32_f16 per K-step.
__global__ __launch_bounds__(256)
void gemm_wmma(const _Float16* __restrict__ xq, const _Float16* __restrict__ wq,
               const float* __restrict__ bias, float* __restrict__ out) {
  __shared__ _Float16 As[2][BM * LDSS];   // 2 x 10240 B
  __shared__ _Float16 Bs[2][BN * LDSS];   // 2 x 10240 B

  const int tid  = threadIdx.x;
  const int lane = tid & 31;
  const int wid  = tid >> 5;           // 0..7
  const int wm   = wid >> 2;           // 0..1  (M direction)
  const int wn   = wid & 3;            // 0..3  (N direction)
  const int m0   = blockIdx.y * BM;
  const int n0   = blockIdx.x * BN;
  const int lrow = lane & 15;          // row/col within fragment
  const int lhi  = lane >> 4;          // which K-half group this lane holds
  const int NT   = DIN / BK;           // 64 K-steps

  // Prologue: DMA k-tile 0 into buffer 0 (wave 0 -> A, wave 1 -> B).
  if (wid == 0)
    tdm_load_tile(xq + (size_t)m0 * DIN, (unsigned)(size_t)(void*)&As[0][0], MTOT);
  if (wid == 1)
    tdm_load_tile(wq + (size_t)n0 * DIN, (unsigned)(size_t)(void*)&Bs[0][0], DOUT);

  v8f acc[4][2] = {};

  for (int kt = 0; kt < NT; ++kt) {
    if (wid < 2) __builtin_amdgcn_s_wait_tensorcnt(0);  // tile kt landed in LDS
    __syncthreads();   // publish buf[kt&1]; all reads of buf[(kt-1)&1] are done

    if (kt + 1 < NT) {                                  // DMA next tile now
      const int nb = (kt + 1) & 1;
      if (wid == 0)
        tdm_load_tile(xq + (size_t)m0 * DIN + (kt + 1) * BK,
                      (unsigned)(size_t)(void*)&As[nb][0], MTOT);
      if (wid == 1)
        tdm_load_tile(wq + (size_t)n0 * DIN + (kt + 1) * BK,
                      (unsigned)(size_t)(void*)&Bs[nb][0], DOUT);
    }

    const _Float16* Ab = &As[kt & 1][0];
    const _Float16* Bb = &Bs[kt & 1][0];

    // B fragments: lane l holds col N=l%16, K = 8*(l/16) + (h%8) + 16*(h/8)
    Frag16 bfr[2];
#pragma unroll
    for (int tn = 0; tn < 2; ++tn) {
      const _Float16* p = Bb + (wn * 32 + tn * 16 + lrow) * LDSS + lhi * 8;
      bfr[tn].h[0] = *(const v8h*)(p);
      bfr[tn].h[1] = *(const v8h*)(p + 16);
    }
    // A fragments: lane l holds row M=l%16, K = 8*(l/16) + (h%8) + 16*(h/8)
    Frag16 afr[4];
#pragma unroll
    for (int tm = 0; tm < 4; ++tm) {
      const _Float16* p = Ab + (wm * 64 + tm * 16 + lrow) * LDSS + lhi * 8;
      afr[tm].h[0] = *(const v8h*)(p);
      afr[tm].h[1] = *(const v8h*)(p + 16);
    }
#pragma unroll
    for (int tm = 0; tm < 4; ++tm)
#pragma unroll
      for (int tn = 0; tn < 2; ++tn)
        acc[tm][tn] = __builtin_amdgcn_wmma_f32_16x16x32_f16(
            false, afr[tm].v, false, bfr[tn].v, (short)0, acc[tm][tn], false, false);
  }

  // Epilogue: C/D layout -> VGPR r holds (M = r + 8*(l/16), N = l%16)
#pragma unroll
  for (int tn = 0; tn < 2; ++tn) {
    int n = n0 + wn * 32 + tn * 16 + lrow;
    float bv = bias[n];
#pragma unroll
    for (int tm = 0; tm < 4; ++tm) {
      int mbase = m0 + wm * 64 + tm * 16 + lhi * 8;
#pragma unroll
      for (int r = 0; r < 8; ++r) {
        out[(size_t)(mbase + r) * DOUT + n] = acc[tm][tn][r] + bv;
      }
    }
  }
}

// ---------------------------------------------------------------------------
extern "C" void kernel_launch(void* const* d_in, const int* in_sizes, int n_in,
                              void* d_out, int out_size, void* d_ws, size_t ws_size,
                              hipStream_t stream) {
  const float* x          = (const float*)d_in[0];
  const float* weight     = (const float*)d_in[1];
  const float* bias       = (const float*)d_in[2];
  const float* log_act_s  = (const float*)d_in[3];
  const float* log_act_q  = (const float*)d_in[4];
  const float* act_b      = (const float*)d_in[5];
  const float* log_wght_s = (const float*)d_in[6];

  _Float16* xq = (_Float16*)d_ws;                                   // 32 MiB
  _Float16* wq = (_Float16*)((char*)d_ws + (size_t)MTOT * DIN * 2); // 32 MiB

  quant_act<<<2048, 256, 0, stream>>>(x, xq, log_act_s, log_act_q, act_b, MTOT * DIN);
  quant_wgt<<<DOUT, 256, 0, stream>>>(weight, wq, log_wght_s);

  dim3 grid(DOUT / BN, MTOT / BM);     // 64 x 64 blocks
  gemm_wmma<<<grid, 256, 0, stream>>>(xq, wq, bias, (float*)d_out);
}
